// Attention_712964571585
// MI455X (gfx1250) — matrix operand order repaired and checked by reading, not compiled, not run on hardware
//
#include <hip/hip_runtime.h>
#include <hip/hip_bf16.h>

// Problem constants (match reference)
#define BB 4
#define GG 2
#define NN 2048
#define DD 512
#define HH 8
#define DHD 64
#define DI 512
#define BGN (BB*GG)
#define NVALID 2049          // 1 null key + 2048 keys
#define NP 2080              // padded to 65 * 32 key blocks
#define QSCALE 0.125f        // dh^-0.5 = 1/8

#define KSTR 72              // LDS K row stride (64 + 8 bf16 pad = 32+4 DW)
#define VSTR 40              // LDS V row stride (32 + 8 bf16 pad = 16+4 DW)

typedef __attribute__((ext_vector_type(16))) __bf16 v16bf;
typedef __attribute__((ext_vector_type(8)))  float  v8f;

// ---- Tensor Data Mover availability probe -------------------------------
#ifndef TRY_TDM
#define TRY_TDM 1
#endif
#if TRY_TDM && defined(__has_builtin)
#  if __has_builtin(__builtin_amdgcn_tensor_load_to_lds)
#    define HAVE_TDM 1
#  endif
#endif
#ifndef HAVE_TDM
#  define HAVE_TDM 0
#endif

__device__ __forceinline__ __bf16 f2bf(float f) { return (__bf16)f; }
__device__ __forceinline__ float  bf2f(__bf16 h) { return (float)h; }

// Reductions across each 16-lane half of the wave (masks < 16 stay inside a half)
__device__ __forceinline__ float red_max16(float v) {
  v = fmaxf(v, __shfl_xor(v, 1, 32));
  v = fmaxf(v, __shfl_xor(v, 2, 32));
  v = fmaxf(v, __shfl_xor(v, 4, 32));
  v = fmaxf(v, __shfl_xor(v, 8, 32));
  return v;
}
__device__ __forceinline__ float red_sum16(float v) {
  v += __shfl_xor(v, 1, 32);
  v += __shfl_xor(v, 2, 32);
  v += __shfl_xor(v, 4, 32);
  v += __shfl_xor(v, 8, 32);
  return v;
}

#if HAVE_TDM
typedef __attribute__((ext_vector_type(4))) unsigned int u32x4;
typedef __attribute__((ext_vector_type(4))) int          i32x4;
typedef __attribute__((ext_vector_type(8))) int          i32x8;

// Build a 2-D TDM descriptor (D#) per CDNA5 ISA and issue
// TENSOR_LOAD_TO_LDS.  data_size = 2 bytes (bf16).  LDS row padding enabled:
// pad_int_code selects DWORDs-per-row (0=2,1=4,2=8,3=16,4=32), pad_amt_code is
// pad DWORDs minus one.
__device__ __forceinline__ void tdm_load_2d(unsigned lds_off, const void* gsrc,
                                            unsigned d0, unsigned d1,
                                            unsigned long long stride0_elems,
                                            unsigned pad_int_code,
                                            unsigned pad_amt_code)
{
  const unsigned long long ga = (unsigned long long)(size_t)gsrc;
  u32x4 g0 = {};
  g0.x = 1u;                                                   // count=1, user D#
  g0.y = lds_off;                                              // lds_addr (bytes)
  g0.z = (unsigned)ga;                                         // global_addr[31:0]
  g0.w = (unsigned)((ga >> 32) & 0x1FFFFFFull) | (2u << 30);   // [56:32] | type=2
  i32x8 g1 = {};
  g1[0] = (int)((1u << 16)                 // data_size = 2B
              | (1u << 20)                 // pad_enable
              | (pad_int_code << 22)
              | (pad_amt_code << 25));
  g1[1] = (int)(d0 << 16);                 // tensor_dim0[15:0] (bits 63:48)
  g1[2] = (int)(d1 << 16);                 // tensor_dim0 hi=0 | tensor_dim1[15:0]
  g1[3] = (int)(d0 << 16);                 // tensor_dim1 hi=0 | tile_dim0
  g1[4] = (int)(d1 & 0xFFFF);              // tile_dim1 | tile_dim2=0
  g1[5] = (int)(stride0_elems & 0xFFFFFFFFull);   // tensor_dim0_stride[31:0]
  g1[6] = (int)((stride0_elems >> 32) & 0xFFFFull);
  g1[7] = 0;
  i32x4 gz = {};
#if __clang_major__ >= 23
  i32x8 gz8 = {};
  __builtin_amdgcn_tensor_load_to_lds(g0, g1, gz, gz, gz8, 0);
#else
  __builtin_amdgcn_tensor_load_to_lds(g0, g1, gz, gz, 0);
#endif
}
#endif // HAVE_TDM

// ---------------------------------------------------------------------------
// Kernel 1: fused Q/K/V grouped projection.  One wave = one 16x16 C tile.
// et in [0,32): Q columns, [32,64): K columns, [64,96): V columns.
// ---------------------------------------------------------------------------
__global__ __launch_bounds__(32)
void proj_kernel(const float* __restrict__ x, const float* __restrict__ Wq,
                 const float* __restrict__ Wkv,
                 __bf16* __restrict__ Qh, __bf16* __restrict__ Kh,
                 __bf16* __restrict__ Vt)
{
  int t = blockIdx.x;
  const int et = t % 96; t /= 96;
  const int nt = t % (NN/16); t /= (NN/16);
  const int bg = t;
  const int g  = bg % GG;

  const int lane = threadIdx.x;
  const int half = lane >> 4;
  const int m    = lane & 15;

  const float* wbase;
  int sec, el0;
  if (et < 32)      { sec = 0; el0 = et*16;       wbase = Wq  + (size_t)g*DI*DD; }
  else if (et < 64) { sec = 1; el0 = (et-32)*16;  wbase = Wkv + (size_t)g*2*DI*DD; }
  else              { sec = 2; el0 = (et-64)*16;  wbase = Wkv + (size_t)g*2*DI*DD + (size_t)DI*DD; }

  const int n = nt*16 + m;
  const float* arow = x + ((size_t)bg*NN + n)*DD;          // A row (M = m)
  const float* brow = wbase + (size_t)(el0 + m)*DD;        // B column (N = m)

  v8f c = {};
  for (int kk = 0; kk < DD; kk += 32) {
    v16bf a, b;
    const int ka0 = kk + half*16;
    const int ka1 = kk + half*16 + 8;
    #pragma unroll
    for (int j = 0; j < 8; ++j) {
      a[j]   = f2bf(arow[ka0 + j]);
      a[8+j] = f2bf(arow[ka1 + j]);
    }
    const int kb = kk + half*16;
    #pragma unroll
    for (int j = 0; j < 16; ++j) b[j] = f2bf(brow[kb + j]);
    c = __builtin_amdgcn_wmma_f32_16x16x32_bf16(false, a, false, b,
                                                (short)0, c, false, false);
  }

  // Branch once per wave (section is uniform); clean strided stores inside.
  const int el   = el0 + m;
  const int h    = el >> 6;
  const int dh   = el & 63;
  const int bgh  = bg*HH + h;
  const int row0 = nt*16 + half*8;           // C-layout: M=r / M=r+8 per half
  if (sec == 0) {
    __bf16* dst = Qh + ((size_t)bgh*NN + row0)*DHD + dh;
    #pragma unroll
    for (int r = 0; r < 8; ++r) dst[(size_t)r*DHD] = f2bf(c[r]);
  } else if (sec == 1) {
    __bf16* dst = Kh + ((size_t)bgh*NP + row0 + 1)*DHD + dh;   // +1: null row 0
    #pragma unroll
    for (int r = 0; r < 8; ++r) dst[(size_t)r*DHD] = f2bf(c[r]);
  } else {
    __bf16* dst = Vt + ((size_t)bgh*DHD + dh)*NP + row0 + 1;   // V transposed
    #pragma unroll
    for (int r = 0; r < 8; ++r) dst[r] = f2bf(c[r]);
  }
}

// ---------------------------------------------------------------------------
// Kernel 2: rotary embedding on Q (with 1/sqrt(dh) scale folded in) and K.
// Each thread owns a (dh, dh+32) pair; ang[n,p] == ang[n,p+32].
// ---------------------------------------------------------------------------
__global__ void rope_kernel(__bf16* __restrict__ Qh, __bf16* __restrict__ Kh,
                            const float* __restrict__ rot_q,
                            const float* __restrict__ rot_k)
{
  long long i = (long long)blockIdx.x * blockDim.x + threadIdx.x;
  const long long QP = (long long)BGN * HH * NN * 32;
  if (i >= 2*QP) return;
  const bool isq = (i < QP);
  long long j = isq ? i : i - QP;
  const int p = (int)(j & 31); j >>= 5;
  const int n = (int)(j % NN); j /= NN;
  const int bgh = (int)j;

  const float ang = isq ? rot_q[n*DHD + p] : rot_k[n*DHD + p];
  const float cs = cosf(ang), sn = sinf(ang);
  __bf16* base = isq ? (Qh + ((size_t)bgh*NN + n)*DHD)
                     : (Kh + ((size_t)bgh*NP + n + 1)*DHD);
  const float t0 = bf2f(base[p]);
  const float t1 = bf2f(base[p + 32]);
  float o0 = t0*cs - t1*sn;
  float o1 = t1*cs + t0*sn;
  if (isq) { o0 *= QSCALE; o1 *= QSCALE; }
  base[p]      = f2bf(o0);
  base[p + 32] = f2bf(o1);
}

// ---------------------------------------------------------------------------
// Kernel 3: write null K/V at key index 0, zero the pad rows (2049..2079).
// ---------------------------------------------------------------------------
__global__ void nullpad_kernel(__bf16* __restrict__ Kh, __bf16* __restrict__ Vt,
                               const float* __restrict__ null_kv)
{
  const int NULLN = BGN*HH*DHD;                  // 4096
  const int PADR  = NP - NVALID;                 // 31
  const int PADN  = BGN*HH*PADR*DHD;
  int i = blockIdx.x * blockDim.x + threadIdx.x;
  if (i < NULLN) {
    const int dh  = i & 63;
    const int bgh = i >> 6;
    const int bg  = bgh >> 3;
    const int h   = bgh & 7;
    const int g   = bg % GG;
    Kh[((size_t)bgh*NP + 0)*DHD + dh] =
        f2bf(null_kv[((size_t)(0*GG + g)*HH + h)*DHD + dh]);
    Vt[((size_t)bgh*DHD + dh)*NP + 0] =
        f2bf(null_kv[((size_t)(1*GG + g)*HH + h)*DHD + dh]);
  } else if (i < NULLN + PADN) {
    int j = i - NULLN;
    const int dh  = j & 63; j >>= 6;
    const int row = NVALID + (j % PADR); j /= PADR;
    const int bgh = j;
    Kh[((size_t)bgh*NP + row)*DHD + dh] = f2bf(0.0f);
    Vt[((size_t)bgh*DHD + dh)*NP + row] = f2bf(0.0f);
  }
}

// ---------------------------------------------------------------------------
// Kernel 4: flash attention.  Block = 4 waves sharing one (b,g,h); each wave
// owns 16 queries.  Per 32-key block, K (32x64) and V^T (64x32) are staged into
// LDS once (TDM tensor_load_to_lds when available, cooperative b128 copies
// otherwise) and consumed by all 4 waves: 4x global-traffic reduction for K/V.
// S = Q.K^T (4 wmma), online softmax via half-wave shuffles, P transposed
// C->A layout through per-wave LDS, O += P.V (4 wmma).
// ---------------------------------------------------------------------------
__global__ __launch_bounds__(128)
void attn_kernel(const __bf16* __restrict__ Qh, const __bf16* __restrict__ Kh,
                 const __bf16* __restrict__ Vt, __bf16* __restrict__ Oh)
{
  __shared__ __bf16 Kbuf[32][KSTR];       // 32 keys x 64 dh (+pad)
  __shared__ __bf16 Vbuf[DHD][VSTR];      // 64 dh  x 32 keys (+pad)
  __shared__ float  plds[4][16][34];      // per-wave P transpose buffer

  int t = blockIdx.x;
  const int qg  = t % (NN/64); t /= (NN/64);
  const int bgh = t;
  const int bg  = bgh >> 3;
  const int h   = bgh & 7;

  const int tid  = threadIdx.x;
  const int w    = tid >> 5;              // wave id 0..3
  const int lane = tid & 31;
  const int half = lane >> 4;
  const int m    = lane & 15;
  const int qbase = qg*64 + w*16;

#if HAVE_TDM
  const unsigned kbufOff = (unsigned)(size_t)(void*)&Kbuf[0][0];
  const unsigned vbufOff = (unsigned)(size_t)(void*)&Vbuf[0][0];
#endif

  // Q A-tiles (dh 0..31 and 32..63), loop-invariant
  const __bf16* qrow = Qh + ((size_t)bgh*NN + qbase + m)*DHD;
  v16bf qa[2];
  #pragma unroll
  for (int kk = 0; kk < 2; ++kk) {
    const int ka0 = kk*32 + half*16, ka1 = kk*32 + half*16 + 8;
    #pragma unroll
    for (int j = 0; j < 8; ++j) { qa[kk][j] = qrow[ka0+j]; qa[kk][8+j] = qrow[ka1+j]; }
  }

  v8f o0 = {}, o1 = {}, o2 = {}, o3 = {};
  float mr[8], lr[8];
  #pragma unroll
  for (int r = 0; r < 8; ++r) { mr[r] = -1e30f; lr[r] = 0.0f; }

  for (int kb = 0; kb < NP; kb += 32) {
    __syncthreads();   // all waves done reading previous K/V block (and plds)

#if HAVE_TDM
    if (w == 0) {
      // K tile: 32 rows of 64 bf16, row stride 64 elems; LDS pad 8 bf16/row
      tdm_load_2d(kbufOff, Kh + ((size_t)bgh*NP + kb)*DHD,
                  /*d0=*/64, /*d1=*/32, /*stride0=*/64ull,
                  /*pad_int=*/4, /*pad_amt=*/3);
      // V tile: 64 rows of 32 bf16, row stride NP elems; LDS pad 8 bf16/row
      tdm_load_2d(vbufOff, Vt + (size_t)bgh*DHD*NP + kb,
                  /*d0=*/32, /*d1=*/64, /*stride0=*/(unsigned long long)NP,
                  /*pad_int=*/3, /*pad_amt=*/3);
      __builtin_amdgcn_s_wait_tensorcnt(0);
    }
#else
    {
      // cooperative staging: 128 threads x 32B each per tensor
      const int key = tid >> 2, segk = tid & 3;                 // K: 32 x 4 segs
      const uint4* ks = (const uint4*)(Kh + ((size_t)bgh*NP + kb + key)*DHD + segk*16);
      uint4* kd = (uint4*)&Kbuf[key][segk*16];
      kd[0] = ks[0]; kd[1] = ks[1];

      const int dh = tid >> 1, segv = tid & 1;                  // V: 64 x 2 segs
      const uint4* vs = (const uint4*)(Vt + ((size_t)bgh*DHD + dh)*NP + kb + segv*16);
      uint4* vd = (uint4*)&Vbuf[dh][segv*16];
      vd[0] = vs[0]; vd[1] = vs[1];

      if (kb + 32 < NP) {                                        // warm next block
        __builtin_prefetch(Kh + ((size_t)bgh*NP + kb + 32 + key)*DHD, 0, 3);
        __builtin_prefetch(Vt + ((size_t)bgh*DHD + dh)*NP + kb + 32, 0, 3);
      }
    }
#endif
    __syncthreads();   // staged K/V visible to all waves

    // S = Q.K^T for two 16-key sub-tiles (B-operand from LDS)
    float s[2][8];
    #pragma unroll
    for (int s2 = 0; s2 < 2; ++s2) {
      const __bf16* krow = &Kbuf[s2*16 + m][0];
      v8f sc = {};
      #pragma unroll
      for (int kk = 0; kk < 2; ++kk) {
        v16bf b;
        const int kbb = kk*32 + half*16;
        #pragma unroll
        for (int j = 0; j < 16; ++j) b[j] = krow[kbb + j];
        sc = __builtin_amdgcn_wmma_f32_16x16x32_bf16(false, qa[kk], false, b,
                                                     (short)0, sc, false, false);
      }
      const bool valid = (kb + s2*16 + m) < NVALID;
      #pragma unroll
      for (int r = 0; r < 8; ++r) s[s2][r] = valid ? sc[r] : -1e30f;
    }

    // online softmax update (per row, reduced across the 16-lane half)
    #pragma unroll
    for (int r = 0; r < 8; ++r) {
      const float bm    = red_max16(fmaxf(s[0][r], s[1][r]));
      const float mnew  = fmaxf(mr[r], bm);
      const float alpha = __expf(mr[r] - mnew);
      const float p0 = __expf(s[0][r] - mnew);
      const float p1 = __expf(s[1][r] - mnew);
      lr[r] = lr[r]*alpha + red_sum16(p0 + p1);
      mr[r] = mnew;
      o0[r] *= alpha; o1[r] *= alpha; o2[r] *= alpha; o3[r] *= alpha;
      plds[w][r + half*8][m]      = p0;          // C-layout -> row-major LDS
      plds[w][r + half*8][16 + m] = p1;
    }
    __syncthreads();

    // reload P in A-layout (16x32 bf16)
    v16bf pa;
    {
      const int ka0 = half*16, ka1 = half*16 + 8;
      #pragma unroll
      for (int j = 0; j < 8; ++j) {
        pa[j]   = f2bf(plds[w][m][ka0 + j]);
        pa[8+j] = f2bf(plds[w][m][ka1 + j]);
      }
    }

    // O(16x64) += P(16x32) x V(32x64), 4 column tiles (B-operand from LDS)
    #pragma unroll
    for (int dt = 0; dt < 4; ++dt) {
      const __bf16* vrow = &Vbuf[dt*16 + m][half*16];
      v16bf b;
      #pragma unroll
      for (int j = 0; j < 16; ++j) b[j] = vrow[j];
      v8f* oacc = (dt == 0) ? &o0 : (dt == 1) ? &o1 : (dt == 2) ? &o2 : &o3;
      *oacc = __builtin_amdgcn_wmma_f32_16x16x32_bf16(false, pa, false, b,
                                                      (short)0, *oacc, false, false);
    }
  }

  // normalize and write merged-head output Oh[bg, n, h*64 + dh] (bf16)
  #pragma unroll
  for (int r = 0; r < 8; ++r) {
    const int n = qbase + r + half*8;
    const float inv = 1.0f / lr[r];
    __bf16* orow = Oh + ((size_t)bg*NN + n)*DI + h*DHD + m;
    orow[0]  = f2bf(o0[r]*inv);
    orow[16] = f2bf(o1[r]*inv);
    orow[32] = f2bf(o2[r]*inv);
    orow[48] = f2bf(o3[r]*inv);
  }
}

// ---------------------------------------------------------------------------
// Kernel 5: output projection out[bg,n,d] = sum_e Oh[bg,n,e] * Wout[g,d,e]
// ---------------------------------------------------------------------------
__global__ __launch_bounds__(32)
void outproj_kernel(const __bf16* __restrict__ Oh, const float* __restrict__ Wout,
                    float* __restrict__ out)
{
  int t = blockIdx.x;
  const int dt = t % 32; t /= 32;
  const int nt = t % (NN/16); t /= (NN/16);
  const int bg = t;
  const int g  = bg % GG;
  const int lane = threadIdx.x, half = lane >> 4, m = lane & 15;

  const __bf16* arow = Oh + ((size_t)bg*NN + nt*16 + m)*DI;
  const float*  brow = Wout + ((size_t)g*DD + dt*16 + m)*DI;

  v8f c = {};
  for (int kk = 0; kk < DI; kk += 32) {
    v16bf a, b;
    const int ka0 = kk + half*16, ka1 = kk + half*16 + 8;
    #pragma unroll
    for (int j = 0; j < 8; ++j) { a[j] = arow[ka0+j]; a[8+j] = arow[ka1+j]; }
    const int kb = kk + half*16;
    #pragma unroll
    for (int j = 0; j < 16; ++j) b[j] = f2bf(brow[kb + j]);
    c = __builtin_amdgcn_wmma_f32_16x16x32_bf16(false, a, false, b,
                                                (short)0, c, false, false);
  }
  #pragma unroll
  for (int r = 0; r < 8; ++r) {
    const int n = nt*16 + r + half*8;
    out[((size_t)bg*NN + n)*DD + dt*16 + m] = c[r];
  }
}

// ---------------------------------------------------------------------------
extern "C" void kernel_launch(void* const* d_in, const int* in_sizes, int n_in,
                              void* d_out, int out_size, void* d_ws, size_t ws_size,
                              hipStream_t stream)
{
  (void)in_sizes; (void)n_in; (void)out_size; (void)ws_size;
  const float* x      = (const float*)d_in[0];
  const float* Wq     = (const float*)d_in[1];
  const float* Wkv    = (const float*)d_in[2];
  const float* Wout   = (const float*)d_in[3];
  const float* nullkv = (const float*)d_in[4];
  const float* rot_q  = (const float*)d_in[5];
  const float* rot_k  = (const float*)d_in[6];
  float* out = (float*)d_out;

  // workspace carve-out (bf16 tensors), ~68 MB total
  char* ws = (char*)d_ws;
  const size_t qh_b = (size_t)BGN*HH*NN*DHD*2;   // Qh
  const size_t kh_b = (size_t)BGN*HH*NP*DHD*2;   // Kh (padded keys)
  const size_t vt_b = kh_b;                      // Vt (transposed, padded)
  __bf16* Qh = (__bf16*)ws;  ws += qh_b;
  __bf16* Kh = (__bf16*)ws;  ws += kh_b;
  __bf16* Vt = (__bf16*)ws;  ws += vt_b;
  __bf16* Oh = (__bf16*)ws;

  proj_kernel<<<BGN*(NN/16)*96, 32, 0, stream>>>(x, Wq, Wkv, Qh, Kh, Vt);

  {
    const long long total = 2LL * BGN * HH * NN * 32;
    rope_kernel<<<(int)((total + 255) / 256), 256, 0, stream>>>(Qh, Kh, rot_q, rot_k);
  }
  {
    const int total = BGN*HH*DHD + BGN*HH*(NP - NVALID)*DHD;
    nullpad_kernel<<<(total + 255) / 256, 256, 0, stream>>>(Kh, Vt, nullkv);
  }

  attn_kernel<<<BGN*HH*(NN/64), 128, 0, stream>>>(Qh, Kh, Vt, Oh);
  outproj_kernel<<<BGN*(NN/16)*32, 32, 0, stream>>>(Oh, Wout, out);
}
